// ComplexMultiHeadAttention_37306085933315
// MI455X (gfx1250) — matrix operand order repaired and checked by reading, not compiled, not run on hardware
//
#include <hip/hip_runtime.h>
#include <hip/hip_bf16.h>
#include <math.h>

// Problem constants (match reference)
#define Bc  4
#define Sc  1024
#define Dc  1024
#define Hc  16
#define DKc 64
#define Mc  (Bc * Sc)          // 4096 rows for the big GEMMs

typedef __attribute__((ext_vector_type(16))) _Float16 v16h;
typedef __attribute__((ext_vector_type(8)))  float    v8f;
typedef unsigned int u32;
typedef __attribute__((ext_vector_type(4))) u32 v4u;
typedef __attribute__((ext_vector_type(8))) int v8i;
typedef __attribute__((ext_vector_type(4))) int v4i;

#define HAVE_TDM __has_builtin(__builtin_amdgcn_tensor_load_to_lds)

// ---------------------------------------------------------------------------
// WMMA wrapper: D = A(16x32 f16) * B(32x16 f16) + C(16x16 f32)
// ---------------------------------------------------------------------------
static __device__ __forceinline__ v8f wmma16(v16h a, v16h b, v8f c) {
  return __builtin_amdgcn_wmma_f32_16x16x32_f16(
      /*neg_a=*/false, a, /*neg_b=*/false, b,
      /*c_mod=*/(short)0, c, /*reuse_a=*/false, /*reuse_b=*/false);
}

// ---------------------------------------------------------------------------
// LDS staging (256-thread cooperative), f32/f16 -> f16 with multiply (VALU path)
// ---------------------------------------------------------------------------
template <typename T>
static __device__ __forceinline__ void stage_tile(_Float16* dst, int dld,
                                                  const T* __restrict__ src,
                                                  int sld, int rows, int cols,
                                                  float mul) {
  for (int idx = threadIdx.x; idx < rows * cols; idx += 256) {
    int r = idx / cols, c = idx - r * cols;
    dst[r * dld + c] = (_Float16)(mul * (float)src[(size_t)r * sld + c]);
  }
}

// ---------------------------------------------------------------------------
// CDNA5 per-lane async staging: GLOBAL_LOAD_ASYNC_TO_LDS_B128 (ASYNCcnt).
// 16B per lane directly into LDS, no VGPR round trip.
// ---------------------------------------------------------------------------
static __device__ __forceinline__ void stage_async_h8(_Float16* dst, int dld,
                                                      const _Float16* __restrict__ src,
                                                      int sld, int rows, int colsH) {
  const int cpr = colsH >> 3;  // b128 chunks per row
  for (int idx = threadIdx.x; idx < rows * cpr; idx += 256) {
    int r = idx / cpr, c = idx - r * cpr;
    unsigned ldsAddr = (unsigned)(size_t)(dst + r * dld + c * 8);
    unsigned long long gAddr =
        (unsigned long long)(size_t)(src + (size_t)r * sld + c * 8);
    asm volatile("global_load_async_to_lds_b128 %0, %1, off"
                 :
                 : "v"(ldsAddr), "v"(gAddr)
                 : "memory");
  }
}

static __device__ __forceinline__ void wait_async_all() {
  asm volatile("s_wait_asynccnt 0x0" ::: "memory");
}

// ---------------------------------------------------------------------------
// CDNA5 Tensor Data Mover: one wave issues a whole 2D f16 tile global->LDS,
// with hardware LDS row padding.  D# packed per CDNA5 ISA.
//   tile_d0   : row length in elements (f16)
//   tile_d1   : number of rows
//   src_ld    : source leading dimension in elements
//   pad_itv   : pad_interval code (3 => 16 DW = 64B, 4 => 32 DW = 128B of data
//               before each pad insertion)
//   pad_amt   : pad_amount code (code+1 DWORDs; 3 => 4 DW = 16B)
// ---------------------------------------------------------------------------
#if HAVE_TDM
static __device__ __forceinline__ void tdm_load_2d(_Float16* ldst,
                                                   const _Float16* gsrc,
                                                   u32 tile_d0, u32 tile_d1,
                                                   u32 src_ld,
                                                   u32 pad_itv, u32 pad_amt) {
  const unsigned long long ga = (unsigned long long)(size_t)gsrc;
  const u32 lds = (u32)(size_t)ldst;

  v4u g0;
  g0[0] = 1u;                                  // count=1, no gather, user D#
  g0[1] = lds;                                 // lds_addr (bytes)
  g0[2] = (u32)ga;                             // global_addr[31:0]
  g0[3] = (u32)(ga >> 32) | 0x80000000u;       // global_addr[56:32] | type=2

  v8i g1;
  g1[0] = (int)((1u << 16)                     // data_size = 1 (2 bytes)
                | (1u << 20)                   // pad_enable
                | (pad_itv << 22)              // pad_interval
                | (pad_amt << 25));            // pad_amount
  g1[1] = (int)((tile_d0 & 0xFFFFu) << 16);    // tensor_dim0[15:0]  (== tile w)
  g1[2] = (int)((tile_d1 & 0xFFFFu) << 16);    // tensor_dim0[31:16]=0, tensor_dim1[15:0]
  g1[3] = (int)((tile_d0 & 0xFFFFu) << 16);    // tensor_dim1[31:16]=0, tile_dim0
  g1[4] = (int)(tile_d1 & 0xFFFFu);            // tile_dim1, tile_dim2=0
  g1[5] = (int)src_ld;                         // tensor_dim0_stride[31:0]
  g1[6] = 0;                                   // dim0_stride[47:32], dim1_stride[15:0]
  g1[7] = 0;                                   // tensor_dim1_stride[47:16]

  const v4i z4 = {0, 0, 0, 0};                 // 2D tensor: groups 2/3 unused
  const v8i z8 = {0, 0, 0, 0, 0, 0, 0, 0};
  __builtin_amdgcn_tensor_load_to_lds(g0, g1, z4, z4, z8, 0);
}

static __device__ __forceinline__ void wait_tensor_all() {
  __builtin_amdgcn_s_wait_tensorcnt((unsigned short)0);
}
#endif

// ---------------------------------------------------------------------------
// Fragment loaders per CDNA5 ISA VGPR layouts (wave32).
// A (16x32, MxK): lanes 0-15 -> M=lane, K base 0; lanes 16-31 -> K base 8.
//   VGPR0..3 = K kb..kb+7, VGPR4..7 = K kb+16..kb+23.
// B (32x16, KxN) from an LDS tile stored [N][K]:
//   lanes 0-15 -> N=lane, K=0..7,16..23 ; lanes 16-31 -> N=lane-16, K=8..15,24..31.
// ---------------------------------------------------------------------------
static __device__ __forceinline__ v16h load_a_frag(const _Float16* lds, int ld,
                                                   int mBase, int lane) {
  const _Float16* p = lds + (mBase + (lane & 15)) * ld + ((lane < 16) ? 0 : 8);
  v16h a;
#pragma unroll
  for (int j = 0; j < 8; ++j) a[j] = p[j];
#pragma unroll
  for (int j = 0; j < 8; ++j) a[8 + j] = p[16 + j];
  return a;
}

static __device__ __forceinline__ v16h load_b_frag(const _Float16* lds, int ld,
                                                   int nBase, int lane) {
  const _Float16* p = lds + (nBase + (lane & 15)) * ld + ((lane < 16) ? 0 : 8);
  v16h b;
#pragma unroll
  for (int j = 0; j < 8; ++j) b[j] = p[j];
#pragma unroll
  for (int j = 0; j < 8; ++j) b[8 + j] = p[16 + j];
  return b;
}

// ---------------------------------------------------------------------------
// Kernel 1: complex projection GEMM.  Y = (Xr+iXi) @ (Wr+iWi)^T.
// Epilogue modes:
//   vmode==0 : head-major [B,H,S,DK], optional negated-imag plane Yni
//   vmode==1 : transposed per-head [B,H,DK,S] (for V, enables DMA AV staging)
// Block tile 128x64, 8 waves (4x2), wave 32x32.
// ---------------------------------------------------------------------------
__global__ void __launch_bounds__(256)
cproj_kernel(const float* __restrict__ Xr, const float* __restrict__ Xi,
             const float* __restrict__ Wr, const float* __restrict__ Wi,
             _Float16* __restrict__ Yr, _Float16* __restrict__ Yi,
             _Float16* __restrict__ Yni, int vmode, float scale) {
  __shared__ _Float16 lXr[128 * 40], lXi[128 * 40];
  __shared__ _Float16 lWr[64 * 40], lWi[64 * 40], lWni[64 * 40];

  const int m0 = blockIdx.y * 128;
  const int n0 = blockIdx.x * 64;
  const int lane = threadIdx.x & 31;
  const int wave = threadIdx.x >> 5;
  const int wm = wave >> 1;       // 0..3
  const int wn = wave & 1;        // 0..1

  v8f cr[2][2] = {};
  v8f ci[2][2] = {};

  for (int k0 = 0; k0 < Dc; k0 += 32) {
    __syncthreads();
    stage_tile(lXr, 40, Xr + (size_t)m0 * Dc + k0, Dc, 128, 32, 1.0f);
    stage_tile(lXi, 40, Xi + (size_t)m0 * Dc + k0, Dc, 128, 32, 1.0f);
    stage_tile(lWr, 40, Wr + (size_t)n0 * Dc + k0, Dc, 64, 32, 1.0f);
    stage_tile(lWi, 40, Wi + (size_t)n0 * Dc + k0, Dc, 64, 32, 1.0f);
    stage_tile(lWni, 40, Wi + (size_t)n0 * Dc + k0, Dc, 64, 32, -1.0f);
    if (k0 + 32 < Dc) {  // hint next K tile toward L2 (global_prefetch_b8)
      __builtin_prefetch(Xr + (size_t)(m0 + (threadIdx.x >> 1)) * Dc + k0 + 32, 0, 1);
    }
    __syncthreads();

    v16h ar[2], ai[2], br[2], bi[2], bni[2];
#pragma unroll
    for (int i = 0; i < 2; ++i) {
      ar[i] = load_a_frag(lXr, 40, wm * 32 + i * 16, lane);
      ai[i] = load_a_frag(lXi, 40, wm * 32 + i * 16, lane);
    }
#pragma unroll
    for (int j = 0; j < 2; ++j) {
      br[j]  = load_b_frag(lWr, 40, wn * 32 + j * 16, lane);
      bi[j]  = load_b_frag(lWi, 40, wn * 32 + j * 16, lane);
      bni[j] = load_b_frag(lWni, 40, wn * 32 + j * 16, lane);
    }
#pragma unroll
    for (int i = 0; i < 2; ++i)
#pragma unroll
      for (int j = 0; j < 2; ++j) {
        cr[i][j] = wmma16(ar[i], br[j], cr[i][j]);   // + Xr.Wr
        cr[i][j] = wmma16(ai[i], bni[j], cr[i][j]);  // - Xi.Wi
        ci[i][j] = wmma16(ai[i], br[j], ci[i][j]);   // + Xi.Wr
        ci[i][j] = wmma16(ar[i], bi[j], ci[i][j]);   // + Xr.Wi
      }
  }

#pragma unroll
  for (int i = 0; i < 2; ++i)
#pragma unroll
    for (int j = 0; j < 2; ++j) {
      const int gmBase = m0 + wm * 32 + i * 16 + ((lane < 16) ? 0 : 8);
      const int gn = n0 + wn * 32 + j * 16 + (lane & 15);
      const int h = gn / DKc, dk = gn & (DKc - 1);
#pragma unroll
      for (int v = 0; v < 8; ++v) {
        const int gm = gmBase + v;
        const int b = gm / Sc, s = gm & (Sc - 1);
        const float vr = cr[i][j][v] * scale;
        const float vi = ci[i][j][v] * scale;
        if (vmode == 0) {
          const size_t off = ((size_t)(b * Hc + h) * Sc + s) * DKc + dk;
          Yr[off] = (_Float16)vr;
          Yi[off] = (_Float16)vi;
          if (Yni) Yni[off] = (_Float16)(-vi);
        } else {  // transposed [B,H,DK,S]
          const size_t off = ((size_t)(b * Hc + h) * DKc + dk) * Sc + s;
          Yr[off] = (_Float16)vr;
          Yi[off] = (_Float16)vi;
        }
      }
    }
}

// ---------------------------------------------------------------------------
// Kernel 2: scores. Per (b,h): ar = qr.kr^T + qi.ki^T ; ai = qi.kr^T - qr.ki^T.
// All 5 tiles are pure f16 copies -> staged by the Tensor Data Mover (one
// descriptor per tile, waves 0..4 each issue one; HW pads 128B rows to the
// 144B LDS stride).  K = DK = 64 staged fully; block tile 128x64.
// ---------------------------------------------------------------------------
__global__ void __launch_bounds__(256)
scores_kernel(const _Float16* __restrict__ qhr, const _Float16* __restrict__ qhi,
              const _Float16* __restrict__ khr, const _Float16* __restrict__ khi,
              const _Float16* __restrict__ khni, float* __restrict__ attn) {
  __shared__ _Float16 lQr[128 * 72], lQi[128 * 72];
  __shared__ _Float16 lKr[64 * 72], lKi[64 * 72], lKni[64 * 72];

  const int bh = blockIdx.z;
  const int m0 = blockIdx.y * 128;
  const int n0 = blockIdx.x * 64;
  const int lane = threadIdx.x & 31;
  const int wave = threadIdx.x >> 5;
  const int wm = wave >> 1, wn = wave & 1;

  const size_t hbase = (size_t)bh * Sc * DKc;
#if HAVE_TDM
  // 64 f16/row = 128B data; pad_interval code 4 = 32 DWORDs (128B),
  // pad_amount code 3 = 4 DWORDs (16B) -> 144B LDS stride (72 halves).
  if (wave == 0)      tdm_load_2d(lQr, qhr + hbase + (size_t)m0 * DKc, 64, 128, DKc, 4, 3);
  else if (wave == 1) tdm_load_2d(lQi, qhi + hbase + (size_t)m0 * DKc, 64, 128, DKc, 4, 3);
  else if (wave == 2) tdm_load_2d(lKr, khr + hbase + (size_t)n0 * DKc, 64, 64, DKc, 4, 3);
  else if (wave == 3) tdm_load_2d(lKi, khi + hbase + (size_t)n0 * DKc, 64, 64, DKc, 4, 3);
  else if (wave == 4) tdm_load_2d(lKni, khni + hbase + (size_t)n0 * DKc, 64, 64, DKc, 4, 3);
  wait_tensor_all();
#else
  stage_async_h8(lQr, 72, qhr + hbase + (size_t)m0 * DKc, DKc, 128, 64);
  stage_async_h8(lQi, 72, qhi + hbase + (size_t)m0 * DKc, DKc, 128, 64);
  stage_async_h8(lKr, 72, khr + hbase + (size_t)n0 * DKc, DKc, 64, 64);
  stage_async_h8(lKi, 72, khi + hbase + (size_t)n0 * DKc, DKc, 64, 64);
  stage_async_h8(lKni, 72, khni + hbase + (size_t)n0 * DKc, DKc, 64, 64);
  wait_async_all();
#endif
  __syncthreads();

  v8f ar[2][2] = {};
  v8f ai[2][2] = {};
#pragma unroll
  for (int kOff = 0; kOff < 64; kOff += 32) {
    v16h aqr[2], aqi[2], bkr[2], bki[2], bkni[2];
#pragma unroll
    for (int i = 0; i < 2; ++i) {
      aqr[i] = load_a_frag(lQr + kOff, 72, wm * 32 + i * 16, lane);
      aqi[i] = load_a_frag(lQi + kOff, 72, wm * 32 + i * 16, lane);
    }
#pragma unroll
    for (int j = 0; j < 2; ++j) {
      bkr[j]  = load_b_frag(lKr + kOff, 72, wn * 32 + j * 16, lane);
      bki[j]  = load_b_frag(lKi + kOff, 72, wn * 32 + j * 16, lane);
      bkni[j] = load_b_frag(lKni + kOff, 72, wn * 32 + j * 16, lane);
    }
#pragma unroll
    for (int i = 0; i < 2; ++i)
#pragma unroll
      for (int j = 0; j < 2; ++j) {
        ar[i][j] = wmma16(aqr[i], bkr[j], ar[i][j]);   // + qr.kr
        ar[i][j] = wmma16(aqi[i], bki[j], ar[i][j]);   // + qi.ki
        ai[i][j] = wmma16(aqi[i], bkr[j], ai[i][j]);   // + qi.kr
        ai[i][j] = wmma16(aqr[i], bkni[j], ai[i][j]);  // - qr.ki
      }
  }

#pragma unroll
  for (int i = 0; i < 2; ++i)
#pragma unroll
    for (int j = 0; j < 2; ++j) {
      const int gmBase = m0 + wm * 32 + i * 16 + ((lane < 16) ? 0 : 8);
      const int gn = n0 + wn * 32 + j * 16 + (lane & 15);
#pragma unroll
      for (int v = 0; v < 8; ++v) {
        const int gm = gmBase + v;
        const float r = ar[i][j][v], im = ai[i][j][v];
        attn[((size_t)bh * Sc + gm) * Sc + gn] = sqrtf(r * r + im * im);
      }
    }
}

// ---------------------------------------------------------------------------
// Kernel 3: stable row softmax in-place over attn rows (length S=1024).
// ---------------------------------------------------------------------------
__global__ void __launch_bounds__(256)
softmax_kernel(float* __restrict__ attn) {
  __shared__ float red[256];
  const size_t row = blockIdx.x;
  float* p = attn + row * Sc;
  const int tid = threadIdx.x;

  float v[4];
  float mx = -3.0e38f;
#pragma unroll
  for (int j = 0; j < 4; ++j) {
    v[j] = p[tid + j * 256];
    mx = fmaxf(mx, v[j]);
  }
  red[tid] = mx;
  __syncthreads();
  for (int s = 128; s > 0; s >>= 1) {
    if (tid < s) red[tid] = fmaxf(red[tid], red[tid + s]);
    __syncthreads();
  }
  mx = red[0];
  __syncthreads();

  float sum = 0.f;
#pragma unroll
  for (int j = 0; j < 4; ++j) {
    v[j] = __expf(v[j] - mx);
    sum += v[j];
  }
  red[tid] = sum;
  __syncthreads();
  for (int s = 128; s > 0; s >>= 1) {
    if (tid < s) red[tid] += red[tid + s];
    __syncthreads();
  }
  const float inv = 1.0f / red[0];
#pragma unroll
  for (int j = 0; j < 4; ++j) p[tid + j * 256] = v[j] * inv;
}

// ---------------------------------------------------------------------------
// Kernel 4: O = attn @ V (real & imag). attn f32 staged with f32->f16 VALU
// conversion; V tiles come from the transposed [B,H,DK,S] plane and are
// staged by the TDM (waves 0/1) each K-step.  Wave = 16 rows x 64 cols.
// Output f16 merged layout [B,S,D].
// ---------------------------------------------------------------------------
__global__ void __launch_bounds__(256)
av_kernel(const float* __restrict__ attn,
          const _Float16* __restrict__ vTr, const _Float16* __restrict__ vTi,
          _Float16* __restrict__ Or, _Float16* __restrict__ Oi) {
  __shared__ _Float16 lA[128 * 40];
  __shared__ _Float16 lVr[64 * 40], lVi[64 * 40];

  const int bh = blockIdx.z;
  const int b = bh / Hc, h = bh & (Hc - 1);
  const int m0 = blockIdx.y * 128;
  const int lane = threadIdx.x & 31;
  const int wave = threadIdx.x >> 5;  // 0..7 -> wave rows

  const _Float16* vTrh = vTr + (size_t)bh * DKc * Sc;
  const _Float16* vTih = vTi + (size_t)bh * DKc * Sc;

  v8f or_[4] = {};
  v8f oi_[4] = {};

  for (int k0 = 0; k0 < Sc; k0 += 32) {
    __syncthreads();
    // V tiles: pure copy [DK][S] slab -> lds [N=64][K=32]
#if HAVE_TDM
    // 32 f16/row = 64B data; pad_interval code 3 = 16 DWORDs (64B),
    // pad_amount code 3 = 4 DWORDs (16B) -> 80B LDS stride (40 halves).
    if (wave == 0)      tdm_load_2d(lVr, vTrh + k0, 32, 64, Sc, 3, 3);
    else if (wave == 1) tdm_load_2d(lVi, vTih + k0, 32, 64, Sc, 3, 3);
#else
    stage_async_h8(lVr, 40, vTrh + k0, Sc, 64, 32);
    stage_async_h8(lVi, 40, vTih + k0, Sc, 64, 32);
#endif
    // attn tile: f32 -> f16 conversion in staging
    stage_tile(lA, 40, attn + ((size_t)bh * Sc + m0) * Sc + k0, Sc, 128, 32, 1.0f);
#if HAVE_TDM
    wait_tensor_all();
#else
    wait_async_all();
#endif
    __syncthreads();

    v16h a = load_a_frag(lA, 40, wave * 16, lane);
#pragma unroll
    for (int j = 0; j < 4; ++j) {
      v16h bvr = load_b_frag(lVr, 40, j * 16, lane);
      v16h bvi = load_b_frag(lVi, 40, j * 16, lane);
      or_[j] = wmma16(a, bvr, or_[j]);
      oi_[j] = wmma16(a, bvi, oi_[j]);
    }
  }

#pragma unroll
  for (int j = 0; j < 4; ++j) {
    const int gmBase = m0 + wave * 16 + ((lane < 16) ? 0 : 8);
    const int dk = j * 16 + (lane & 15);
#pragma unroll
    for (int v = 0; v < 8; ++v) {
      const int s = gmBase + v;
      const size_t off = ((size_t)b * Sc + s) * Dc + h * DKc + dk;
      Or[off] = (_Float16)or_[j][v];
      Oi[off] = (_Float16)oi_[j][v];
    }
  }
}

// ---------------------------------------------------------------------------
// Kernel 5: FC complex GEMM + residual (+ dropout epsilon on imag), f32 out.
// X f16 [M,D]; W f32 [D,D]; xr = fr + res_r ; xi = fi + res_i + 1e-10.
// X staging keeps the per-lane GLOBAL_LOAD_ASYNC_TO_LDS_B128 path; W staging
// converts f32 -> f16 (VALU).
// ---------------------------------------------------------------------------
__global__ void __launch_bounds__(256)
cfc_kernel(const _Float16* __restrict__ Xr, const _Float16* __restrict__ Xi,
           const float* __restrict__ Wr, const float* __restrict__ Wi,
           const float* __restrict__ resR, const float* __restrict__ resI,
           float* __restrict__ outR, float* __restrict__ outI) {
  __shared__ _Float16 lXr[128 * 40], lXi[128 * 40];
  __shared__ _Float16 lWr[64 * 40], lWi[64 * 40], lWni[64 * 40];

  const int m0 = blockIdx.y * 128;
  const int n0 = blockIdx.x * 64;
  const int lane = threadIdx.x & 31;
  const int wave = threadIdx.x >> 5;
  const int wm = wave >> 1, wn = wave & 1;

  v8f cr[2][2] = {};
  v8f ci[2][2] = {};

  for (int k0 = 0; k0 < Dc; k0 += 32) {
    __syncthreads();
    stage_async_h8(lXr, 40, Xr + (size_t)m0 * Dc + k0, Dc, 128, 32);
    stage_async_h8(lXi, 40, Xi + (size_t)m0 * Dc + k0, Dc, 128, 32);
    stage_tile(lWr, 40, Wr + (size_t)n0 * Dc + k0, Dc, 64, 32, 1.0f);
    stage_tile(lWi, 40, Wi + (size_t)n0 * Dc + k0, Dc, 64, 32, 1.0f);
    stage_tile(lWni, 40, Wi + (size_t)n0 * Dc + k0, Dc, 64, 32, -1.0f);
    wait_async_all();
    __syncthreads();

    v16h ar[2], ai[2], br[2], bi[2], bni[2];
#pragma unroll
    for (int i = 0; i < 2; ++i) {
      ar[i] = load_a_frag(lXr, 40, wm * 32 + i * 16, lane);
      ai[i] = load_a_frag(lXi, 40, wm * 32 + i * 16, lane);
    }
#pragma unroll
    for (int j = 0; j < 2; ++j) {
      br[j]  = load_b_frag(lWr, 40, wn * 32 + j * 16, lane);
      bi[j]  = load_b_frag(lWi, 40, wn * 32 + j * 16, lane);
      bni[j] = load_b_frag(lWni, 40, wn * 32 + j * 16, lane);
    }
#pragma unroll
    for (int i = 0; i < 2; ++i)
#pragma unroll
      for (int j = 0; j < 2; ++j) {
        cr[i][j] = wmma16(ar[i], br[j], cr[i][j]);
        cr[i][j] = wmma16(ai[i], bni[j], cr[i][j]);
        ci[i][j] = wmma16(ai[i], br[j], ci[i][j]);
        ci[i][j] = wmma16(ar[i], bi[j], ci[i][j]);
      }
  }

#pragma unroll
  for (int i = 0; i < 2; ++i)
#pragma unroll
    for (int j = 0; j < 2; ++j) {
      const int gmBase = m0 + wm * 32 + i * 16 + ((lane < 16) ? 0 : 8);
      const int gn = n0 + wn * 32 + j * 16 + (lane & 15);
#pragma unroll
      for (int v = 0; v < 8; ++v) {
        const int gm = gmBase + v;
        const size_t off = (size_t)gm * Dc + gn;
        outR[off] = cr[i][j][v] + resR[off];
        outI[off] = ci[i][j][v] + resI[off] + 1e-10f;
      }
    }
}

// ---------------------------------------------------------------------------
// Kernels 6/7: deterministic two-stage per-batch mean over (S,D) of xr & xi.
// ---------------------------------------------------------------------------
__global__ void __launch_bounds__(256)
batch_partial_kernel(const float* __restrict__ xr, const float* __restrict__ xi,
                     float* __restrict__ partR, float* __restrict__ partI) {
  __shared__ float rr[256], ri[256];
  const int b = blockIdx.y;
  const int blk = blockIdx.x;  // 256 blocks per batch, 4096 elems each
  const size_t base = (size_t)b * Sc * Dc + (size_t)blk * 4096;
  const int tid = threadIdx.x;
  float sr = 0.f, si = 0.f;
#pragma unroll
  for (int j = 0; j < 16; ++j) {
    const int id = tid + j * 256;
    sr += xr[base + id];
    si += xi[base + id];
  }
  rr[tid] = sr;
  ri[tid] = si;
  __syncthreads();
  for (int s = 128; s > 0; s >>= 1) {
    if (tid < s) { rr[tid] += rr[tid + s]; ri[tid] += ri[tid + s]; }
    __syncthreads();
  }
  if (tid == 0) {
    partR[b * 256 + blk] = rr[0];
    partI[b * 256 + blk] = ri[0];
  }
}

__global__ void __launch_bounds__(256)
batch_final_kernel(const float* __restrict__ partR, const float* __restrict__ partI,
                   float* __restrict__ sums) {
  __shared__ float rr[256], ri[256];
  const int b = blockIdx.x;
  const int tid = threadIdx.x;
  rr[tid] = partR[b * 256 + tid];
  ri[tid] = partI[b * 256 + tid];
  __syncthreads();
  for (int s = 128; s > 0; s >>= 1) {
    if (tid < s) { rr[tid] += rr[tid + s]; ri[tid] += ri[tid + s]; }
    __syncthreads();
  }
  if (tid == 0) {
    const float inv = 1.0f / ((float)Sc * (float)Dc);
    sums[b] = rr[0] * inv;       // mean real
    sums[4 + b] = ri[0] * inv;   // mean imag
  }
}

// ---------------------------------------------------------------------------
// Kernel 8: complex LayerNorm finalize -> yr, yi in d_out.
// ---------------------------------------------------------------------------
__global__ void __launch_bounds__(256)
finalize_kernel(const float* __restrict__ xr, const float* __restrict__ xi,
                const float* __restrict__ sums,
                const float* __restrict__ gamma, const float* __restrict__ beta,
                float* __restrict__ yr, float* __restrict__ yi) {
  const size_t i = (size_t)blockIdx.x * 256 + threadIdx.x;
  const int b = (int)(i / ((size_t)Sc * Dc));
  const int d = (int)(i & (Dc - 1));
  const float cr = xr[i] - sums[b];
  const float ci = xi[i] - sums[4 + b];
  const float istd = rsqrtf(cr * cr + ci * ci + 1e-6f);
  const float g = gamma[d];
  yr[i] = g * cr * istd + beta[d];
  yi[i] = g * ci * istd;
}

// ---------------------------------------------------------------------------
// Host launcher
// ---------------------------------------------------------------------------
extern "C" void kernel_launch(void* const* d_in, const int* in_sizes, int n_in,
                              void* d_out, int out_size, void* d_ws, size_t ws_size,
                              hipStream_t stream) {
  (void)in_sizes; (void)n_in; (void)out_size; (void)ws_size;
  const float* q_r   = (const float*)d_in[0];
  const float* q_i   = (const float*)d_in[1];
  const float* k_r   = (const float*)d_in[2];
  const float* k_i   = (const float*)d_in[3];
  const float* v_r   = (const float*)d_in[4];
  const float* v_i   = (const float*)d_in[5];
  const float* wq_r  = (const float*)d_in[6];
  const float* wq_i  = (const float*)d_in[7];
  const float* wk_r  = (const float*)d_in[8];
  const float* wk_i  = (const float*)d_in[9];
  const float* wv_r  = (const float*)d_in[10];
  const float* wv_i  = (const float*)d_in[11];
  const float* wfc_r = (const float*)d_in[12];
  const float* wfc_i = (const float*)d_in[13];
  const float* gamma = (const float*)d_in[14];
  const float* beta  = (const float*)d_in[15];

  // Workspace layout
  char* ws = (char*)d_ws;
  const size_t NE = (size_t)Bc * Sc * Dc;        // 4M elements
  const size_t H2 = NE * sizeof(_Float16);       // 8 MB per f16 plane
  _Float16* qhr  = (_Float16*)(ws + 0 * H2);
  _Float16* qhi  = (_Float16*)(ws + 1 * H2);
  _Float16* khr  = (_Float16*)(ws + 2 * H2);
  _Float16* khi  = (_Float16*)(ws + 3 * H2);
  _Float16* khni = (_Float16*)(ws + 4 * H2);     // negated k imag plane
  _Float16* vTr  = (_Float16*)(ws + 5 * H2);     // V transposed [B,H,DK,S]
  _Float16* vTi  = (_Float16*)(ws + 6 * H2);
  _Float16* ohr  = (_Float16*)(ws + 7 * H2);
  _Float16* ohi  = (_Float16*)(ws + 8 * H2);
  float* xr    = (float*)(ws + 9 * H2);
  float* xi    = (float*)(ws + 9 * H2 + NE * sizeof(float));
  float* partR = (float*)(ws + 9 * H2 + 2 * NE * sizeof(float));
  float* partI = partR + Bc * 256;
  float* sums  = partI + Bc * 256;

  // Output layout: yr [B,S,D], yi [B,S,D], attn [B,H,S,S]
  float* yr   = (float*)d_out;
  float* yi   = yr + NE;
  float* attn = yr + 2 * NE;

  const dim3 blk(256);
  const dim3 gGemm(Dc / 64, Mc / 128);           // 16 x 32

  // 1) complex projections (q scaled by 1/sqrt(DK) = 0.125)
  cproj_kernel<<<gGemm, blk, 0, stream>>>(q_r, q_i, wq_r, wq_i, qhr, qhi,
                                          nullptr, 0, 0.125f);
  cproj_kernel<<<gGemm, blk, 0, stream>>>(k_r, k_i, wk_r, wk_i, khr, khi,
                                          khni, 0, 1.0f);
  cproj_kernel<<<gGemm, blk, 0, stream>>>(v_r, v_i, wv_r, wv_i, vTr, vTi,
                                          nullptr, 1, 1.0f);

  // 2) attention score magnitudes (pre-softmax) straight into attn output
  scores_kernel<<<dim3(Sc / 64, Sc / 128, Bc * Hc), blk, 0, stream>>>(
      qhr, qhi, khr, khi, khni, attn);

  // 3) softmax in-place (this IS the attn output)
  softmax_kernel<<<dim3(Bc * Hc * Sc), blk, 0, stream>>>(attn);

  // 4) O = attn @ V
  av_kernel<<<dim3(1, Sc / 128, Bc * Hc), blk, 0, stream>>>(attn, vTr, vTi, ohr, ohi);

  // 5) FC + residual + dropout epsilon
  cfc_kernel<<<gGemm, blk, 0, stream>>>(ohr, ohi, wfc_r, wfc_i, q_r, q_i, xr, xi);

  // 6-7) per-batch means (deterministic two-stage reduction)
  batch_partial_kernel<<<dim3(256, Bc), blk, 0, stream>>>(xr, xi, partR, partI);
  batch_final_kernel<<<dim3(Bc), blk, 0, stream>>>(partR, partI, sums);

  // 8) complex LayerNorm
  finalize_kernel<<<dim3((unsigned)(NE / 256)), blk, 0, stream>>>(
      xr, xi, sums, gamma, beta, yr, yi);
}
